// GraphFormation_9113920602710
// MI455X (gfx1250) — matrix operand order repaired and checked by reading, not compiled, not run on hardware
//
#include <hip/hip_runtime.h>

typedef _Float16 v16h __attribute__((ext_vector_type(16)));
typedef _Float16 v8h  __attribute__((ext_vector_type(8)));
typedef float    v8f  __attribute__((ext_vector_type(8)));

#define N_NODES  100000
#define DEGREE   32
#define N_PAIRS  8192
#define FEAT_DIM 9
#define HID      20
#define OUT      15

#define WAVES_PER_BLOCK 10   // 10 waves * 16 nodes = 160 nodes/block; 625 blocks = 100000
#define TILE            16

// ---------------------------------------------------------------------------
// Kernel 1: precompute h1[v] = relu(W1 @ concat(feat[v], mean(feat[adj[v]])))
// for ALL nodes (dedup: 540k demanded enc1 evals -> 100k unique).
// One wave per 16-node tile; GEMM via V_WMMA_F32_16X16X32_F16 (K padded 18->32,
// two N-tiles cover HID=20). Fragments per CDNA5 ISA 7.12.2 wave32 layouts,
// loaded as packed ds_load_b128 chunks.
// ---------------------------------------------------------------------------
__global__ __launch_bounds__(WAVES_PER_BLOCK * 32)
void enc1_h1_kernel(const int* __restrict__ adj,
                    const float* __restrict__ feat,
                    const float* __restrict__ W1,
                    float* __restrict__ h1)
{
    // X tiles: [wave][node][K] f16, K padded to 32 (rows 64B-aligned).
    __shared__ _Float16 xtile[WAVES_PER_BLOCK][TILE][32];
    // W1 padded, stored by output neuron: wpad[n][k] (rows 64B-aligned) so a
    // B-fragment (K 0-7 & 16-23, or 8-15 & 24-31, of one output column) lies in one row.
    __shared__ _Float16 wpad[32][32];

    const int tid = threadIdx.x;
    const int w   = tid >> 5;   // wave id in block
    const int L   = tid & 31;   // lane id (wave32)

    // Stage padded W1 once per block: wpad[n][k] = W1[n][k] (20x18 live).
    for (int idx = tid; idx < 32 * 32; idx += blockDim.x) {
        const int n = idx >> 5, k = idx & 31;
        const float val = (n < HID && k < 2 * FEAT_DIM) ? W1[n * (2 * FEAT_DIM) + k] : 0.0f;
        wpad[n][k] = (_Float16)val;
    }

    const int tile = blockIdx.x * WAVES_PER_BLOCK + w;
    const int base = tile * TILE;          // first node of this wave's tile
    const int m    = L >> 1;               // node-in-tile (2 lanes per node)
    const int half = L & 1;                // which 16 of the 32 neighbors
    const int node = base + m;

    // --- Gather: each lane sums 16 neighbor feature rows (L2-resident). ---
    // adj row chunk: 16 contiguous ints, 64B-aligned -> 4x global_load_b128.
    float acc[FEAT_DIM];
    #pragma unroll
    for (int c = 0; c < FEAT_DIM; ++c) acc[c] = 0.0f;

    const int4* arow4 = (const int4*)(adj + (long)node * DEGREE + half * 16);
    #pragma unroll
    for (int q = 0; q < 4; ++q) {
        const int4 nb4 = arow4[q];
        const int nbs[4] = {nb4.x, nb4.y, nb4.z, nb4.w};
        #pragma unroll
        for (int u = 0; u < 4; ++u) {
            const float* fr = feat + (long)nbs[u] * FEAT_DIM;  // 36B rows: scalar loads
            #pragma unroll
            for (int c = 0; c < FEAT_DIM; ++c) acc[c] += fr[c];
        }
    }

    float selff[FEAT_DIM];
    const float* sr = feat + (long)node * FEAT_DIM;
    #pragma unroll
    for (int c = 0; c < FEAT_DIM; ++c) selff[c] = sr[c];

    // Combine the two half-sums within the lane pair (wave32 shuffle).
    #pragma unroll
    for (int c = 0; c < FEAT_DIM; ++c) {
        const float other = __shfl_xor(acc[c], 1, 32);
        acc[c] = (acc[c] + other) * (1.0f / (float)DEGREE);
    }

    // Lane with half==0 writes the node's padded X row: 4x ds_store_b128.
    if (half == 0) {
        _Float16 xr[32];
        #pragma unroll
        for (int k = 0; k < 32; ++k) {
            float xv = 0.0f;
            if (k < FEAT_DIM)          xv = selff[k];
            else if (k < 2 * FEAT_DIM) xv = acc[k - FEAT_DIM];
            xr[k] = (_Float16)xv;
        }
        v8h* dst = (v8h*)&xtile[w][m][0];
        #pragma unroll
        for (int c = 0; c < 4; ++c) {
            v8h t;
            #pragma unroll
            for (int i = 0; i < 8; ++i) t[i] = xr[c * 8 + i];
            dst[c] = t;
        }
    }
    __syncthreads();

    // --- A fragment (16-bit A 16x32): lanes 0-15 -> K 0-7 & 16-23 of row M=L;
    //     lanes 16-31 -> K 8-15 & 24-31 of row M=L-16.
    //     Each half is one 16B-aligned ds_load_b128 chunk.
    const int am   = L & 15;
    const int achk = (L >= 16) ? 1 : 0;            // chunk 0: K0-7, chunk 1: K8-15
    const v8h* xr8 = (const v8h*)&xtile[w][am][0];
    const v8h alo  = xr8[achk];                    // K 0-7 or 8-15
    const v8h ahi  = xr8[achk + 2];                // K 16-23 or 24-31
    const v16h a   = __builtin_shufflevector(alo, ahi,
                        0, 1, 2, 3, 4, 5, 6, 7, 8, 9, 10, 11, 12, 13, 14, 15);

    // --- B fragments (16-bit B 32x16): lanes 0-15 hold K=0-7 & 16-23 of column N;
    //     lanes 16-31 hold K=8-15 & 24-31. Column N of B = row N of wpad -> two
    //     16B ds_load_b128 chunks per fragment.
    const int bn   = L & 15;
    const int bchk = (L >= 16) ? 1 : 0;            // lanes 0-15 -> chunks 0,2; lanes 16-31 -> 1,3
    const v8h* w0  = (const v8h*)&wpad[bn][0];
    const v8h* w1  = (const v8h*)&wpad[16 + bn][0];
    const v16h b0  = __builtin_shufflevector(w0[bchk], w0[bchk + 2],
                        0, 1, 2, 3, 4, 5, 6, 7, 8, 9, 10, 11, 12, 13, 14, 15);
    const v16h b1  = __builtin_shufflevector(w1[bchk], w1[bchk + 2],
                        0, 1, 2, 3, 4, 5, 6, 7, 8, 9, 10, 11, 12, 13, 14, 15);

    v8f c0 = {}, c1 = {};
    c0 = __builtin_amdgcn_wmma_f32_16x16x32_f16(false, a, false, b0, (short)0, c0, false, false);
    c1 = __builtin_amdgcn_wmma_f32_16x16x32_f16(false, a, false, b1, (short)0, c1, false, false);

    // --- D layout: lane L -> N=L&15; VGPR r -> M=r (+8 for lanes 16-31). ---
    const int rbase = (L >= 16) ? 8 : 0;
    #pragma unroll
    for (int r = 0; r < 8; ++r) {
        float d = c0[r];
        d = d > 0.0f ? d : 0.0f;                        // relu
        h1[(long)(base + rbase + r) * HID + bn] = d;
    }
    if (bn < HID - 16) {                                // hid 16..19
        #pragma unroll
        for (int r = 0; r < 8; ++r) {
            float d = c1[r];
            d = d > 0.0f ? d : 0.0f;
            h1[(long)(base + rbase + r) * HID + 16 + bn] = d;
        }
    }
}

// ---------------------------------------------------------------------------
// Kernel 2: per pair -> pool h1 over (32 neighbors + self) for both endpoints,
// W2 GEMV + relu, concat, fc1 + relu, fc2 + relu. Tiny compute, L2 gathers.
// ---------------------------------------------------------------------------
__global__ __launch_bounds__(64)
void enc2_mlp_kernel(const int* __restrict__ to_pred,
                     const int* __restrict__ adj,
                     const float* __restrict__ h1,
                     const float* __restrict__ W2,
                     const float* __restrict__ fc1_w,
                     const float* __restrict__ fc1_b,
                     const float* __restrict__ fc2_w,
                     const float* __restrict__ fc2_b,
                     float* __restrict__ out)
{
    __shared__ float p[2][HID];
    __shared__ float e[2][OUT];
    __shared__ float hbuf[OUT];

    const int pair = blockIdx.x;
    const int t    = threadIdx.x;

    // Mean-pool h1 over {adj[v], v}: threads 0..39 each own (endpoint, component).
    if (t < 2 * HID) {
        const int ep = t / HID, c = t % HID;
        const int node = to_pred[pair * 2 + ep];
        const int* arow = adj + (long)node * DEGREE;
        float s = h1[(long)node * HID + c];             // self (gcn=True includes self)
        for (int r = 0; r < DEGREE; ++r) {
            const int nb = arow[r];
            s += h1[(long)nb * HID + c];
        }
        p[ep][c] = s * (1.0f / (float)(DEGREE + 1));
    }
    __syncthreads();

    // e = relu(W2 @ p)
    if (t < 2 * OUT) {
        const int ep = t / OUT, o = t % OUT;
        float s = 0.0f;
        #pragma unroll
        for (int k = 0; k < HID; ++k) s += W2[o * HID + k] * p[ep][k];
        e[ep][o] = s > 0.0f ? s : 0.0f;
    }
    __syncthreads();

    // h = relu(fc1 @ concat(e1, e2) + b1)
    if (t < OUT) {
        float s = fc1_b[t];
        #pragma unroll
        for (int k = 0; k < 2 * OUT; ++k) {
            const float ck = (k < OUT) ? e[0][k] : e[1][k - OUT];
            s += fc1_w[t * (2 * OUT) + k] * ck;
        }
        hbuf[t] = s > 0.0f ? s : 0.0f;
    }
    __syncthreads();

    // out = relu(fc2 @ h + b2)
    if (t == 0) {
        float s = fc2_b[0];
        #pragma unroll
        for (int k = 0; k < OUT; ++k) s += fc2_w[k] * hbuf[k];
        out[pair] = s > 0.0f ? s : 0.0f;
    }
}

extern "C" void kernel_launch(void* const* d_in, const int* in_sizes, int n_in,
                              void* d_out, int out_size, void* d_ws, size_t ws_size,
                              hipStream_t stream) {
    (void)in_sizes; (void)n_in; (void)out_size; (void)ws_size;
    const int*   to_pred = (const int*)  d_in[0];
    const int*   adj     = (const int*)  d_in[1];
    const float* feat    = (const float*)d_in[2];
    const float* W1      = (const float*)d_in[3];
    const float* W2      = (const float*)d_in[4];
    const float* fc1_w   = (const float*)d_in[5];
    const float* fc1_b   = (const float*)d_in[6];
    const float* fc2_w   = (const float*)d_in[7];
    const float* fc2_b   = (const float*)d_in[8];
    float* out = (float*)d_out;

    // Workspace: h1 for all nodes = 100000 * 20 * 4B = 8 MB (L2-resident).
    float* h1 = (float*)d_ws;

    const int blocks1 = N_NODES / (WAVES_PER_BLOCK * TILE);   // 625, exact
    enc1_h1_kernel<<<blocks1, WAVES_PER_BLOCK * 32, 0, stream>>>(adj, feat, W1, h1);
    enc2_mlp_kernel<<<N_PAIRS, 64, 0, stream>>>(to_pred, adj, h1, W2,
                                                fc1_w, fc1_b, fc2_w, fc2_b, out);
}
